// GCNLatencyPredictor_28123445854866
// MI455X (gfx1250) — matrix operand order repaired and checked by reading, compiled, not run on hardware
//
#include <hip/hip_runtime.h>
#include <hip/hip_bf16.h>

typedef __attribute__((ext_vector_type(16))) _Float16 v16h;
typedef __attribute__((ext_vector_type(8)))  _Float16 v8h;
typedef __attribute__((ext_vector_type(8)))  float    v8f;
typedef __attribute__((ext_vector_type(4)))  float    v4f;

#define NODE_DIM   12
#define GLOB_DIM   11
#define IN_CH      23    // NODE_DIM + GLOB_DIM
#define HID        64
#define HID2       32
#define NTGT       4
#define K1PAD      160   // 133 padded to 5 chunks of 32
#define EIN_LD     168   // multiple of 8 halves -> 16B-aligned rows, nice banks
#define Z1_LD      72
#define Z2_LD      40
#define FRAG_H     512   // halves per pre-swizzled B fragment (32 lanes x 16)

// ---------------------------------------------------------------- WMMA utils

static __device__ __forceinline__ v8f wmma_f16(v16h a, v16h b, v8f c) {
  // 8 args: (neg_a, A, neg_b, B, c_mod, C, reuse_a, reuse_b)
  return __builtin_amdgcn_wmma_f32_16x16x32_f16(false, a, false, b, (short)0, c,
                                                false, false);
}

// A fragment (16x32 f16) from LDS row-major tile t[row*ld + k].
// Lane L: row = L&15, kbase = 16*(L>>4); element e -> K = kbase + e
// => two contiguous 8-half runs: [kb, kb+8) and [kb+8, kb+16): 2x ds_load_b128
static __device__ __forceinline__ v16h load_a_frag(const _Float16* t, int ld,
                                                   int kc, int lane) {
  const int row = lane & 15;
  const int kb  = (lane >> 4) << 4;
  const _Float16* p = t + row * ld + kc + kb;   // 16B aligned by construction
  v8h lo = *(const v8h*)(p);
  v8h hi = *(const v8h*)(p + 8);
  return __builtin_shufflevector(lo, hi, 0, 1, 2, 3, 4, 5, 6, 7,
                                 8, 9, 10, 11, 12, 13, 14, 15);
}

// B fragment from pre-swizzled LDS: contiguous 32B per lane (2x ds_load_b128)
static __device__ __forceinline__ v16h load_b_packed(const _Float16* base,
                                                     int frag, int lane) {
  return *(const v16h*)(base + frag * FRAG_H + lane * 16);
}

// ---------------------------------------------------------------- GCN kernels

__global__ void k_fill(float* __restrict__ p, float v, int n) {
  int i = blockIdx.x * blockDim.x + threadIdx.x;
  if (i < n) p[i] = v;
}

__global__ void k_deg_count(const int* __restrict__ dst, float* __restrict__ deg,
                            int m) {
  int i = blockIdx.x * blockDim.x + threadIdx.x;
  if (i < m) atomicAdd(&deg[dst[i]], 1.0f);
}

__global__ void k_rsqrt_inplace(float* __restrict__ d, int n) {
  int i = blockIdx.x * blockDim.x + threadIdx.x;
  if (i < n) d[i] = rsqrtf(d[i]);   // deg >= 1 always (self loops)
}

__global__ void k_build_h0(const float* __restrict__ x, const float* __restrict__ u,
                           float* __restrict__ h0, int n) {
  int idx = blockIdx.x * blockDim.x + threadIdx.x;
  if (idx >= n * IN_CH) return;
  int node = idx / IN_CH, c = idx - node * IN_CH;
  h0[idx] = (c < NODE_DIM) ? x[(size_t)node * NODE_DIM + c] : u[c - NODE_DIM];
}

// one node per 64-thread block: out[node][t] = sum_k h[node][k] * W[k][t]
__global__ __launch_bounds__(64) void k_xform(const float* __restrict__ h,
                                              const float* __restrict__ W,
                                              float* __restrict__ out,
                                              int n, int in_dim) {
  __shared__ float row[HID];
  const int node = blockIdx.x;
  const int t = threadIdx.x;
  if (t < in_dim) row[t] = h[(size_t)node * in_dim + t];
  __syncthreads();
  float s = 0.0f;
  for (int k = 0; k < in_dim; ++k) s += row[k] * W[k * HID + t];
  out[(size_t)node * HID + t] = s;
}

// thread per (edge, 4-feature chunk): scatter hw[src]*norm into agg[dst]
__global__ void k_scatter(const int* __restrict__ src, const int* __restrict__ dst,
                          const float* __restrict__ dinv,
                          const float* __restrict__ hw, float* __restrict__ agg,
                          int m) {
  int idx = blockIdx.x * blockDim.x + threadIdx.x;
  if (idx >= m * 16) return;
  int e = idx >> 4, q = idx & 15;
  int s = src[e], d = dst[e];
  float nrm = dinv[s] * dinv[d];
  const float* hp = hw + (size_t)s * HID + q * 4;
  float* ap = agg + (size_t)d * HID + q * 4;
#pragma unroll
  for (int j = 0; j < 4; ++j) atomicAdd(&ap[j], hp[j] * nrm);
}

// add self-loop term + bias, ReLU; result stored back into agg (becomes h)
__global__ void k_finalize(const float* __restrict__ hw, const float* __restrict__ dinv,
                           const float* __restrict__ b, float* __restrict__ agg,
                           int n) {
  int idx = blockIdx.x * blockDim.x + threadIdx.x;
  if (idx >= n * HID) return;
  int node = idx >> 6, f = idx & 63;
  float di = dinv[node];
  float v = agg[idx] + hw[idx] * di * di + b[f];
  agg[idx] = v > 0.0f ? v : 0.0f;
}

// ---------------------------------------------------------------- Edge MLP (WMMA)

__global__ __launch_bounds__(128) void k_edge_mlp(
    const int* __restrict__ src, const int* __restrict__ dst,
    const float* __restrict__ eattr, const float* __restrict__ h2,
    const float* __restrict__ D1, const float* __restrict__ bd1,
    const float* __restrict__ D2, const float* __restrict__ bd2,
    const float* __restrict__ D3, const float* __restrict__ bd3,
    float* __restrict__ out, int M) {
  // B weights pre-swizzled into per-lane fragment layout (frag*512 + lane*16)
  __shared__ __align__(32) _Float16 sW1f[20 * FRAG_H];  // 5 kt x 4 nt
  __shared__ __align__(32) _Float16 sW2f[4 * FRAG_H];   // 2 kt x 2 nt
  __shared__ __align__(32) _Float16 sW3f[1 * FRAG_H];
  __shared__ __align__(32) _Float16 sEin[4 * 16 * EIN_LD];
  __shared__ __align__(32) _Float16 sZ1[4 * 16 * Z1_LD];
  __shared__ __align__(32) _Float16 sZ2[4 * 16 * Z2_LD];

  const int tid  = threadIdx.x;
  const int wave = tid >> 5;
  const int lane = tid & 31;

  // ---- stage weights in fragment layout:
  // within a fragment: lane ln holds col n = nc + (ln&15), K = kc + 16*(ln>>4) + e
  for (int i = tid; i < 20 * FRAG_H; i += blockDim.x) {
    int f = i >> 9, r = i & 511;
    int ln = r >> 4, e = r & 15;
    int kt = f >> 2, nt = f & 3;
    int k = kt * 32 + ((ln >> 4) << 4) + e;
    int n = nt * 16 + (ln & 15);
    sW1f[i] = (_Float16)((k < 2 * HID + 5) ? D1[k * HID + n] : 0.0f);
  }
  for (int i = tid; i < 4 * FRAG_H; i += blockDim.x) {
    int f = i >> 9, r = i & 511;
    int ln = r >> 4, e = r & 15;
    int kt = f >> 1, nt = f & 1;
    int k = kt * 32 + ((ln >> 4) << 4) + e;
    int n = nt * 16 + (ln & 15);
    sW2f[i] = (_Float16)D2[k * HID2 + n];
  }
  for (int i = tid; i < FRAG_H; i += blockDim.x) {
    int ln = i >> 4, e = i & 15;
    int k = ((ln >> 4) << 4) + e;
    int n = ln & 15;
    sW3f[i] = (_Float16)((n < NTGT) ? D3[k * NTGT + n] : 0.0f);
  }
  __syncthreads();

  _Float16* ein = sEin + wave * 16 * EIN_LD;
  _Float16* z1  = sZ1  + wave * 16 * Z1_LD;
  _Float16* z2  = sZ2  + wave * 16 * Z2_LD;

  // ---- hoist biases (depend only on lane)
  const int nlo = lane & 15;
  const int mb  = (lane >> 4) << 3;
  float bn1[4], bn2[2];
#pragma unroll
  for (int nt = 0; nt < 4; ++nt) bn1[nt] = bd1[nlo + nt * 16];
#pragma unroll
  for (int nt = 0; nt < 2; ++nt) bn2[nt] = bd2[nlo + nt * 16];
  const float bn3 = (nlo < NTGT) ? bd3[nlo] : 0.0f;

  const int T = (M + 15) >> 4;               // 16-edge tiles
  // uniform trip count per block -> __syncthreads() is safe in the loop
  for (int base = blockIdx.x * 4; base < T; base += gridDim.x * 4) {
    const int tile = base + wave;
    const bool active = (tile < T);          // wave-uniform

    // ---- stage e_in tile: [h2[src] | h2[dst] | edge_attr | 0-pad] as f16
    {
      const int row  = lane & 15;
      const int half = lane >> 4;
      const int e    = tile * 16 + row;
      _Float16* r = ein + row * EIN_LD;
      if (active && e < M) {
        const int s = src[e], d = dst[e];
        const v4f* hs = (const v4f*)(h2 + (size_t)s * HID);  // 256B-aligned rows
        const v4f* hd = (const v4f*)(h2 + (size_t)d * HID);
#pragma unroll
        for (int b = 0; b < 4; ++b) {        // 4 vector blocks of 8 cols each
          const int c0 = half * 32 + b * 8;
          v4f f0 = hs[c0 >> 2], f1 = hs[(c0 >> 2) + 1];
          v4f g0 = hd[c0 >> 2], g1 = hd[(c0 >> 2) + 1];
          v8h hsrc, hdst;
#pragma unroll
          for (int j = 0; j < 4; ++j) {
            hsrc[j] = (_Float16)f0[j]; hsrc[4 + j] = (_Float16)f1[j];
            hdst[j] = (_Float16)g0[j]; hdst[4 + j] = (_Float16)g1[j];
          }
          *(v8h*)(r + c0)       = hsrc;      // cols [0,64)
          *(v8h*)(r + HID + c0) = hdst;      // cols [64,128)
        }
        if (half == 0) {                     // cols [128,160): attr + zero pad
          const float* ea = eattr + (size_t)e * 5;
          v8h t = {};
#pragma unroll
          for (int j = 0; j < 5; ++j) t[j] = (_Float16)ea[j];
          v8h z = {};
          *(v8h*)(r + 128) = t;
          *(v8h*)(r + 136) = z;
          *(v8h*)(r + 144) = z;
          *(v8h*)(r + 152) = z;
        }
      } else if (active) {                   // pad rows past M: zero
        v8h z = {};
#pragma unroll
        for (int c0 = 0; c0 < 80; c0 += 8) *(v8h*)(r + half * 80 + c0) = z;
      }
    }
    __syncthreads();

    // ---- layer 1: (16x160) @ (160x64), bias+ReLU -> z1 (16x64 f16)
    if (active) {
      v16h a0 = load_a_frag(ein, EIN_LD,   0, lane);
      v16h a1 = load_a_frag(ein, EIN_LD,  32, lane);
      v16h a2 = load_a_frag(ein, EIN_LD,  64, lane);
      v16h a3 = load_a_frag(ein, EIN_LD,  96, lane);
      v16h a4 = load_a_frag(ein, EIN_LD, 128, lane);
#pragma unroll
      for (int nt = 0; nt < 4; ++nt) {
        v8f acc = {};
        acc = wmma_f16(a0, load_b_packed(sW1f,  0 + nt, lane), acc);
        acc = wmma_f16(a1, load_b_packed(sW1f,  4 + nt, lane), acc);
        acc = wmma_f16(a2, load_b_packed(sW1f,  8 + nt, lane), acc);
        acc = wmma_f16(a3, load_b_packed(sW1f, 12 + nt, lane), acc);
        acc = wmma_f16(a4, load_b_packed(sW1f, 16 + nt, lane), acc);
        const int n = nlo + nt * 16;
        const float bn = bn1[nt];
#pragma unroll
        for (int r = 0; r < 8; ++r) {
          float v = acc[r] + bn;
          z1[(mb + r) * Z1_LD + n] = (_Float16)(v > 0.0f ? v : 0.0f);
        }
      }
    }
    __syncthreads();

    // ---- layer 2: (16x64) @ (64x32), bias+ReLU -> z2 (16x32 f16)
    if (active) {
      v16h a0 = load_a_frag(z1, Z1_LD,  0, lane);
      v16h a1 = load_a_frag(z1, Z1_LD, 32, lane);
#pragma unroll
      for (int nt = 0; nt < 2; ++nt) {
        v8f acc = {};
        acc = wmma_f16(a0, load_b_packed(sW2f, 0 + nt, lane), acc);
        acc = wmma_f16(a1, load_b_packed(sW2f, 2 + nt, lane), acc);
        const int n = nlo + nt * 16;
        const float bn = bn2[nt];
#pragma unroll
        for (int r = 0; r < 8; ++r) {
          float v = acc[r] + bn;
          z2[(mb + r) * Z2_LD + n] = (_Float16)(v > 0.0f ? v : 0.0f);
        }
      }
    }
    __syncthreads();

    // ---- layer 3: (16x32) @ (32x16-padded), bias, write out[:, :4]
    if (active) {
      v16h a0 = load_a_frag(z2, Z2_LD, 0, lane);
      v8f acc = {};
      acc = wmma_f16(a0, load_b_packed(sW3f, 0, lane), acc);
      if (nlo < NTGT) {
#pragma unroll
        for (int r = 0; r < 8; ++r) {
          const int e = tile * 16 + mb + r;
          if (e < M) out[(size_t)e * NTGT + nlo] = acc[r] + bn3;
        }
      }
    }
    __syncthreads();
  }
}

// ---------------------------------------------------------------- launch

extern "C" void kernel_launch(void* const* d_in, const int* in_sizes, int n_in,
                              void* d_out, int out_size, void* d_ws, size_t ws_size,
                              hipStream_t stream) {
  (void)n_in; (void)out_size; (void)ws_size;
  const float* x    = (const float*)d_in[0];
  const int*   ei   = (const int*)  d_in[1];
  const float* ea   = (const float*)d_in[2];
  const float* u    = (const float*)d_in[3];
  const float* W1   = (const float*)d_in[4];
  const float* b1   = (const float*)d_in[5];
  const float* W2   = (const float*)d_in[6];
  const float* b2   = (const float*)d_in[7];
  const float* D1   = (const float*)d_in[8];
  const float* bd1  = (const float*)d_in[9];
  const float* D2   = (const float*)d_in[10];
  const float* bd2  = (const float*)d_in[11];
  const float* D3   = (const float*)d_in[12];
  const float* bd3  = (const float*)d_in[13];

  const int N = in_sizes[0] / NODE_DIM;   // 50000
  const int M = in_sizes[1] / 2;          // 800000
  const int* src = ei;
  const int* dst = ei + M;

  float* ws   = (float*)d_ws;
  float* dinv = ws;                               // N
  float* h0   = dinv + N;                         // N*23
  float* bufA = h0 + (size_t)N * IN_CH;           // N*64 (hw)
  float* bufB = bufA + (size_t)N * HID;           // N*64 (agg/h)
  float* outp = (float*)d_out;

  const int TB = 256;
  // degree -> dinv
  k_fill<<<(N + TB - 1) / TB, TB, 0, stream>>>(dinv, 1.0f, N);
  k_deg_count<<<(M + TB - 1) / TB, TB, 0, stream>>>(dst, dinv, M);
  k_rsqrt_inplace<<<(N + TB - 1) / TB, TB, 0, stream>>>(dinv, N);

  // conv 1
  k_build_h0<<<(N * IN_CH + TB - 1) / TB, TB, 0, stream>>>(x, u, h0, N);
  k_xform<<<N, 64, 0, stream>>>(h0, W1, bufA, N, IN_CH);
  k_fill<<<(N * HID + TB - 1) / TB, TB, 0, stream>>>(bufB, 0.0f, N * HID);
  k_scatter<<<(M * 16 + TB - 1) / TB, TB, 0, stream>>>(src, dst, dinv, bufA, bufB, M);
  k_finalize<<<(N * HID + TB - 1) / TB, TB, 0, stream>>>(bufA, dinv, b1, bufB, N);

  // conv 2
  k_xform<<<N, 64, 0, stream>>>(bufB, W2, bufA, N, HID);
  k_fill<<<(N * HID + TB - 1) / TB, TB, 0, stream>>>(bufB, 0.0f, N * HID);
  k_scatter<<<(M * 16 + TB - 1) / TB, TB, 0, stream>>>(src, dst, dinv, bufA, bufB, M);
  k_finalize<<<(N * HID + TB - 1) / TB, TB, 0, stream>>>(bufA, dinv, b2, bufB, N);

  // edge MLP (WMMA) : bufB == h2
  k_edge_mlp<<<1024, 128, 0, stream>>>(src, dst, ea, bufB,
                                       D1, bd1, D2, bd2, D3, bd3, outp, M);
}
